// DNRI_Encoder_56487409877019
// MI455X (gfx1250) — compile-verified, matrix operand order checked by reading, not run on hardware
//
#include <hip/hip_runtime.h>

#define NNODES 30
#define H      64
#define INSZ   8
#define KOUT   2
#define BDIM   16
#define TDIM   96
#define NEDGE  870               // NNODES*(NNODES-1)
#define NSEQ   (BDIM * NEDGE)    // 13920

typedef __attribute__((ext_vector_type(16))) _Float16 v16h;
typedef __attribute__((ext_vector_type(8)))  float    v8f;

// ---------------------------------------------------------------------------
// WMMA helpers (CDNA5 16x16x32 f16 -> f32)
// ---------------------------------------------------------------------------
__device__ __forceinline__ v8f wmma16(v16h a, v16h b, v8f c) {
    // 8 args: (neg_a, A, neg_b, B, c_mod, C, reuse_a, reuse_b)
    return __builtin_amdgcn_wmma_f32_16x16x32_f16(false, a, false, b,
                                                  (short)0, c, false, false);
}

// 16x32 f16 fragment from row-major memory (ISA 7.12.2 16-bit A/B layout):
// lanes 0-15: row = lane, K = {0..7,16..23}; lanes 16-31: row = lane-16,
// K = {8..15,24..31}; two K values packed per VGPR.
__device__ __forceinline__ v16h frag_h(const _Float16* src, int ld) {
    int lane = threadIdx.x & 31;
    int row  = lane & 15;
    int kb   = (lane < 16) ? 0 : 8;
    const _Float16* p = src + (size_t)row * ld;
    v16h a;
#pragma unroll
    for (int j = 0; j < 4; ++j) {
        a[2*j]         = p[kb + 2*j];
        a[2*j + 1]     = p[kb + 2*j + 1];
        a[8 + 2*j]     = p[kb + 16 + 2*j];
        a[8 + 2*j + 1] = p[kb + 16 + 2*j + 1];
    }
    return a;
}

// Same fragment but sourced from f32 (optionally column-bounded + scaled)
__device__ __forceinline__ v16h frag_f(const float* src, int ld, int kmax, float scale) {
    int lane = threadIdx.x & 31;
    int row  = lane & 15;
    int kb   = (lane < 16) ? 0 : 8;
    const float* p = src + (size_t)row * ld;
    v16h a;
#pragma unroll
    for (int j = 0; j < 8; ++j) {
        int k = (j < 4) ? (kb + 2*j) : (kb + 16 + 2*(j - 4));
        a[2*j]     = (_Float16)((k     < kmax) ? p[k]     * scale : 0.f);
        a[2*j + 1] = (_Float16)((k + 1 < kmax) ? p[k + 1] * scale : 0.f);
    }
    return a;
}

// C/D layout: element (m,n): n = lane&15, m = vgpr + (lane>=16 ? 8 : 0).
// Bias column is constant per lane -> single scalar load.
__device__ __forceinline__ v8f add_bias(v8f acc, const float* bias, int n0, bool relu) {
    float b = bias[n0 + (threadIdx.x & 15)];
#pragma unroll
    for (int r = 0; r < 8; ++r) {
        float v = acc[r] + b;
        acc[r] = relu ? fmaxf(v, 0.f) : v;
    }
    return acc;
}

__device__ __forceinline__ void tile_to_h(_Float16* dst, int ld, v8f acc) {
    int lane = threadIdx.x & 31;
    int n  = lane & 15;
    int mb = (lane < 16) ? 0 : 8;
#pragma unroll
    for (int r = 0; r < 8; ++r)
        dst[(size_t)(mb + r) * ld + n] = (_Float16)acc[r];
}

// ---------------------------------------------------------------------------
// Weight f32 -> f16 convert (with optional zero-padding of K dim)
// ---------------------------------------------------------------------------
__global__ void cvt_w_kernel(const float* __restrict__ src, _Float16* __restrict__ dst,
                             int rows, int scols, int dcols) {
    int i = blockIdx.x * blockDim.x + threadIdx.x;
    if (i >= rows * dcols) return;
    int r = i / dcols, c = i % dcols;
    dst[i] = (c < scols) ? (_Float16)src[r * scols + c] : (_Float16)0.f;
}

// ---------------------------------------------------------------------------
// MLP1: [B,T,N,8] f32 -> x1 [B,N,T,64] f16   (Linear 8->64, ReLU, 64->64)
// grid (B*N, T/16), block 128 (4 waves, each owns 16 output cols)
// ---------------------------------------------------------------------------
__global__ void mlp1_kernel(const float* __restrict__ x,
                            const _Float16* __restrict__ W1,  // 64x32 (padded)
                            const float* __restrict__ b1,
                            const _Float16* __restrict__ W2,  // 64x64
                            const float* __restrict__ b2,
                            _Float16* __restrict__ y) {
    int b = blockIdx.x / NNODES, n = blockIdx.x % NNODES;
    int t0 = blockIdx.y * 16;
    int n0 = (threadIdx.x >> 5) * 16;
    __shared__ _Float16 hb[16][H];

    const float* arow = x + ((size_t)(b * TDIM + t0) * NNODES + n) * INSZ;
    v16h a = frag_f(arow, NNODES * INSZ, INSZ, 1.f);
    v8f acc = {};
    acc = wmma16(a, frag_h(W1 + n0 * 32, 32), acc);
    acc = add_bias(acc, b1, n0, true);
    tile_to_h(&hb[0][n0], H, acc);
    __syncthreads();

    v16h h0 = frag_h(&hb[0][0], H), h1 = frag_h(&hb[0][32], H);
    v8f o = {};
    o = wmma16(h0, frag_h(W2 + n0 * H, H), o);
    o = wmma16(h1, frag_h(W2 + n0 * H + 32, H), o);
    o = add_bias(o, b2, n0, false);
    tile_to_h(y + ((size_t)(b * NNODES + n) * TDIM + t0) * H + n0, H, o);
}

// ---------------------------------------------------------------------------
// MLP2: x1 gathered by (send,recv) -> x2 [B,E,T,64] f16 ; fused segment-sum
// (atomicAdd f32) into agg [B,N,T,64] at receiver node.
// ---------------------------------------------------------------------------
__global__ void mlp2_kernel(const _Float16* __restrict__ x1,
                            const int* __restrict__ send, const int* __restrict__ recv,
                            const _Float16* __restrict__ W1,  // 64x128
                            const float* __restrict__ b1,
                            const _Float16* __restrict__ W2,  // 64x64
                            const float* __restrict__ b2,
                            _Float16* __restrict__ x2, float* __restrict__ agg) {
    int b = blockIdx.x / NEDGE, e = blockIdx.x % NEDGE;
    int t0 = blockIdx.y * 16;
    int n0 = (threadIdx.x >> 5) * 16;
    int se = send[e], re = recv[e];
    __shared__ _Float16 hb[16][H];

    const _Float16* xs = x1 + ((size_t)(b * NNODES + se) * TDIM + t0) * H;
    const _Float16* xr = x1 + ((size_t)(b * NNODES + re) * TDIM + t0) * H;
    v16h a0 = frag_h(xs, H), a1 = frag_h(xs + 32, H);
    v16h a2 = frag_h(xr, H), a3 = frag_h(xr + 32, H);
    const _Float16* wb = W1 + n0 * 128;
    v8f acc = {};
    acc = wmma16(a0, frag_h(wb,      128), acc);
    acc = wmma16(a1, frag_h(wb + 32, 128), acc);
    acc = wmma16(a2, frag_h(wb + 64, 128), acc);
    acc = wmma16(a3, frag_h(wb + 96, 128), acc);
    acc = add_bias(acc, b1, n0, true);
    tile_to_h(&hb[0][n0], H, acc);
    __syncthreads();

    v16h h0 = frag_h(&hb[0][0], H), h1 = frag_h(&hb[0][32], H);
    v8f o = {};
    o = wmma16(h0, frag_h(W2 + n0 * H, H), o);
    o = wmma16(h1, frag_h(W2 + n0 * H + 32, H), o);
    o = add_bias(o, b2, n0, false);
    tile_to_h(x2 + ((size_t)(b * NEDGE + e) * TDIM + t0) * H + n0, H, o);

    // edge2node segment sum (normalization folded into MLP3 A-load)
    float* ap = agg + ((size_t)(b * NNODES + re) * TDIM + t0) * H + n0;
    int lane = threadIdx.x & 31;
    int nn = lane & 15, mb = (lane < 16) ? 0 : 8;
#pragma unroll
    for (int r = 0; r < 8; ++r)
        atomicAdd(&ap[(size_t)(mb + r) * H + nn], o[r]);
}

// ---------------------------------------------------------------------------
// MLP3: agg f32 [B,N,T,64] (scaled by 1/29) -> x3 [B,N,T,64] f16
// ---------------------------------------------------------------------------
__global__ void mlp3_kernel(const float* __restrict__ agg,
                            const _Float16* __restrict__ W1, const float* __restrict__ b1,
                            const _Float16* __restrict__ W2, const float* __restrict__ b2,
                            _Float16* __restrict__ y) {
    int b = blockIdx.x / NNODES, n = blockIdx.x % NNODES;
    int t0 = blockIdx.y * 16;
    int n0 = (threadIdx.x >> 5) * 16;
    __shared__ _Float16 hb[16][H];
    const float norm = 1.f / (NNODES - 1);

    const float* arow = agg + ((size_t)(b * NNODES + n) * TDIM + t0) * H;
    v16h a0 = frag_f(arow, H, 32, norm);
    v16h a1 = frag_f(arow + 32, H, 32, norm);
    v8f acc = {};
    acc = wmma16(a0, frag_h(W1 + n0 * H, H), acc);
    acc = wmma16(a1, frag_h(W1 + n0 * H + 32, H), acc);
    acc = add_bias(acc, b1, n0, true);
    tile_to_h(&hb[0][n0], H, acc);
    __syncthreads();

    v16h h0 = frag_h(&hb[0][0], H), h1 = frag_h(&hb[0][32], H);
    v8f o = {};
    o = wmma16(h0, frag_h(W2 + n0 * H, H), o);
    o = wmma16(h1, frag_h(W2 + n0 * H + 32, H), o);
    o = add_bias(o, b2, n0, false);
    tile_to_h(y + ((size_t)(b * NNODES + n) * TDIM + t0) * H + n0, H, o);
}

// ---------------------------------------------------------------------------
// MLP4: concat(x3[send], x3[recv], x2_skip) (K=192) -> xf [B,E,T,64] f16
// ---------------------------------------------------------------------------
__global__ void mlp4_kernel(const _Float16* __restrict__ x3,
                            const _Float16* __restrict__ x2,
                            const int* __restrict__ send, const int* __restrict__ recv,
                            const _Float16* __restrict__ W1,  // 64x192
                            const float* __restrict__ b1,
                            const _Float16* __restrict__ W2,  // 64x64
                            const float* __restrict__ b2,
                            _Float16* __restrict__ xf) {
    int b = blockIdx.x / NEDGE, e = blockIdx.x % NEDGE;
    int t0 = blockIdx.y * 16;
    int n0 = (threadIdx.x >> 5) * 16;
    int se = send[e], re = recv[e];
    __shared__ _Float16 hb[16][H];

    const _Float16* xs = x3 + ((size_t)(b * NNODES + se) * TDIM + t0) * H;
    const _Float16* xr = x3 + ((size_t)(b * NNODES + re) * TDIM + t0) * H;
    const _Float16* xk = x2 + ((size_t)(b * NEDGE  + e)  * TDIM + t0) * H;
    const _Float16* wb = W1 + n0 * 192;
    v8f acc = {};
    acc = wmma16(frag_h(xs,      H), frag_h(wb,       192), acc);
    acc = wmma16(frag_h(xs + 32, H), frag_h(wb + 32,  192), acc);
    acc = wmma16(frag_h(xr,      H), frag_h(wb + 64,  192), acc);
    acc = wmma16(frag_h(xr + 32, H), frag_h(wb + 96,  192), acc);
    acc = wmma16(frag_h(xk,      H), frag_h(wb + 128, 192), acc);
    acc = wmma16(frag_h(xk + 32, H), frag_h(wb + 160, 192), acc);
    acc = add_bias(acc, b1, n0, true);
    tile_to_h(&hb[0][n0], H, acc);
    __syncthreads();

    v16h h0 = frag_h(&hb[0][0], H), h1 = frag_h(&hb[0][32], H);
    v8f o = {};
    o = wmma16(h0, frag_h(W2 + n0 * H, H), o);
    o = wmma16(h1, frag_h(W2 + n0 * H + 32, H), o);
    o = add_bias(o, b2, n0, false);
    tile_to_h(xf + ((size_t)(b * NEDGE + e) * TDIM + t0) * H + n0, H, o);
}

// ---------------------------------------------------------------------------
// Forward GRU: one wave per 16 sequences, loop T. Per step: 12 N-tiles x
// (2 WMMA x@Wih + 2 WMMA h@Whh), gate math in VALU, h in LDS f32.
// Fused: fy store (f16), prior projection (K=2), final hT.
// ---------------------------------------------------------------------------
__global__ void gru_fwd_kernel(const _Float16* __restrict__ xf,
                               const _Float16* __restrict__ WIH,  // 192x64 f16
                               const _Float16* __restrict__ WHH,  // 192x64 f16
                               const float* __restrict__ bih, const float* __restrict__ bhh,
                               const float* __restrict__ pri_w, const float* __restrict__ pri_b,
                               _Float16* __restrict__ fy, float* __restrict__ prior,
                               float* __restrict__ hT) {
    int sb   = blockIdx.x * 16;
    int lane = threadIdx.x & 31;
    __shared__ float hs[16][H];
    __shared__ float gx[16][3 * H];
    __shared__ float gh[16][3 * H];
    for (int i = lane; i < 16 * H; i += 32) hs[i >> 6][i & 63] = 0.f;

    for (int t = 0; t < TDIM; ++t) {
        const _Float16* xrow = xf + ((size_t)sb * TDIM + t) * H;
        v16h ax0 = frag_h(xrow, TDIM * H);
        v16h ax1 = frag_h(xrow + 32, TDIM * H);
        v16h ah0 = frag_f(&hs[0][0], H, 32, 1.f);
        v16h ah1 = frag_f(&hs[0][32], H, 32, 1.f);
#pragma unroll
        for (int nt = 0; nt < 12; ++nt) {
            int n0 = nt * 16;
            v8f aX = {}, aH = {};
            aX = wmma16(ax0, frag_h(WIH + n0 * H, H), aX);
            aX = wmma16(ax1, frag_h(WIH + n0 * H + 32, H), aX);
            aH = wmma16(ah0, frag_h(WHH + n0 * H, H), aH);
            aH = wmma16(ah1, frag_h(WHH + n0 * H + 32, H), aH);
            int nn = lane & 15, mb = (lane < 16) ? 0 : 8;
            float bx = bih[n0 + nn], bh = bhh[n0 + nn];
#pragma unroll
            for (int r = 0; r < 8; ++r) {
                gx[mb + r][n0 + nn] = aX[r] + bx;
                gh[mb + r][n0 + nn] = aH[r] + bh;
            }
        }
        // gate math (torch order r,z,n) + state update + fy store
#pragma unroll 4
        for (int i = 0; i < 32; ++i) {
            int idx = lane * 32 + i;
            int m = idx >> 6, n = idx & 63;
            float rg = 1.f / (1.f + __expf(-(gx[m][n] + gh[m][n])));
            float zg = 1.f / (1.f + __expf(-(gx[m][64 + n] + gh[m][64 + n])));
            float ng = tanhf(gx[m][128 + n] + rg * gh[m][128 + n]);
            float h  = (1.f - zg) * ng + zg * hs[m][n];
            hs[m][n] = h;
            fy[((size_t)(sb + m) * TDIM + t) * H + n] = (_Float16)h;
        }
        // prior = fy @ pri_w.T + pri_b  (K=2): lane -> (m = lane&15, k = lane>>4)
        {
            int m = lane & 15, k = lane >> 4;
            float s = pri_b[k];
            for (int n = 0; n < H; ++n) s += hs[m][n] * pri_w[k * H + n];
            int seq = sb + m, b = seq / NEDGE, e = seq % NEDGE;
            prior[((size_t)(b * TDIM + t) * NEDGE + e) * KOUT + k] = s;
        }
    }
    for (int i = lane; i < 16 * H; i += 32)
        hT[(size_t)(sb + (i >> 6)) * H + (i & 63)] = hs[i >> 6][i & 63];
}

// ---------------------------------------------------------------------------
// Reverse GRU (processes xf[:, ::-1]); posterior fused: at time t the current
// h equals ry[:, ::-1][t], combined with stored fy[t] against enc_w [2,128].
// ---------------------------------------------------------------------------
__global__ void gru_rev_kernel(const _Float16* __restrict__ xf,
                               const _Float16* __restrict__ WIH,
                               const _Float16* __restrict__ WHH,
                               const float* __restrict__ bih, const float* __restrict__ bhh,
                               const float* __restrict__ enc_w, const float* __restrict__ enc_b,
                               const _Float16* __restrict__ fy, float* __restrict__ post) {
    int sb   = blockIdx.x * 16;
    int lane = threadIdx.x & 31;
    __shared__ float hs[16][H];
    __shared__ float gx[16][3 * H];
    __shared__ float gh[16][3 * H];
    for (int i = lane; i < 16 * H; i += 32) hs[i >> 6][i & 63] = 0.f;

    for (int t = TDIM - 1; t >= 0; --t) {
        const _Float16* xrow = xf + ((size_t)sb * TDIM + t) * H;
        v16h ax0 = frag_h(xrow, TDIM * H);
        v16h ax1 = frag_h(xrow + 32, TDIM * H);
        v16h ah0 = frag_f(&hs[0][0], H, 32, 1.f);
        v16h ah1 = frag_f(&hs[0][32], H, 32, 1.f);
#pragma unroll
        for (int nt = 0; nt < 12; ++nt) {
            int n0 = nt * 16;
            v8f aX = {}, aH = {};
            aX = wmma16(ax0, frag_h(WIH + n0 * H, H), aX);
            aX = wmma16(ax1, frag_h(WIH + n0 * H + 32, H), aX);
            aH = wmma16(ah0, frag_h(WHH + n0 * H, H), aH);
            aH = wmma16(ah1, frag_h(WHH + n0 * H + 32, H), aH);
            int nn = lane & 15, mb = (lane < 16) ? 0 : 8;
            float bx = bih[n0 + nn], bh = bhh[n0 + nn];
#pragma unroll
            for (int r = 0; r < 8; ++r) {
                gx[mb + r][n0 + nn] = aX[r] + bx;
                gh[mb + r][n0 + nn] = aH[r] + bh;
            }
        }
#pragma unroll 4
        for (int i = 0; i < 32; ++i) {
            int idx = lane * 32 + i;
            int m = idx >> 6, n = idx & 63;
            float rg = 1.f / (1.f + __expf(-(gx[m][n] + gh[m][n])));
            float zg = 1.f / (1.f + __expf(-(gx[m][64 + n] + gh[m][64 + n])));
            float ng = tanhf(gx[m][128 + n] + rg * gh[m][128 + n]);
            hs[m][n] = (1.f - zg) * ng + zg * hs[m][n];
        }
        // posterior = [fy[t], h] @ enc_w.T + enc_b
        {
            int m = lane & 15, k = lane >> 4;
            const _Float16* frow = fy + ((size_t)(sb + m) * TDIM + t) * H;
            float s = enc_b[k];
            for (int n = 0; n < H; ++n)
                s += (float)frow[n] * enc_w[k * 2 * H + n]
                   + hs[m][n]       * enc_w[k * 2 * H + H + n];
            int seq = sb + m, b = seq / NEDGE, e = seq % NEDGE;
            post[((size_t)(b * TDIM + t) * NEDGE + e) * KOUT + k] = s;
        }
    }
}

// ---------------------------------------------------------------------------
extern "C" void kernel_launch(void* const* d_in, const int* in_sizes, int n_in,
                              void* d_out, int out_size, void* d_ws, size_t ws_size,
                              hipStream_t stream) {
    (void)in_sizes; (void)n_in; (void)out_size; (void)ws_size;

    const float* inputs = (const float*)d_in[0];
    const int*   send   = (const int*)d_in[1];
    const int*   recv   = (const int*)d_in[2];
    const float* b1_1 = (const float*)d_in[4],  *b1_2 = (const float*)d_in[6];
    const float* b2_1 = (const float*)d_in[8],  *b2_2 = (const float*)d_in[10];
    const float* b3_1 = (const float*)d_in[12], *b3_2 = (const float*)d_in[14];
    const float* b4_1 = (const float*)d_in[16], *b4_2 = (const float*)d_in[18];
    const float* f_bih = (const float*)d_in[21], *f_bhh = (const float*)d_in[22];
    const float* r_bih = (const float*)d_in[25], *r_bhh = (const float*)d_in[26];
    const float* enc_w = (const float*)d_in[27], *enc_b = (const float*)d_in[28];
    const float* pri_w = (const float*)d_in[29], *pri_b = (const float*)d_in[30];

    const size_t NODE_E = (size_t)BDIM * NNODES * TDIM * H;   // 2,949,120
    const size_t EDGE_E = (size_t)BDIM * NEDGE  * TDIM * H;   // 85,524,480

    char* ws = (char*)d_ws;
    size_t off = 0;
    _Float16* WGT = (_Float16*)(ws + off); off += 92160 * sizeof(_Float16);
    float*    AGG = (float*)   (ws + off); off += NODE_E * sizeof(float);
    _Float16* X1  = (_Float16*)(ws + off); off += NODE_E * sizeof(_Float16);
    _Float16* X3  = (_Float16*)(ws + off); off += NODE_E * sizeof(_Float16);
    _Float16* X2  = (_Float16*)(ws + off); off += EDGE_E * sizeof(_Float16);
    _Float16* XF  = (_Float16*)(ws + off); off += EDGE_E * sizeof(_Float16);
    _Float16* FY  = (_Float16*)(ws + off); off += EDGE_E * sizeof(_Float16);

    _Float16* W1_1 = WGT;                 // 64x32 (zero-padded from 64x8)
    _Float16* W1_2 = W1_1 + 64 * 32;
    _Float16* W2_1 = W1_2 + 64 * 64;      // 64x128
    _Float16* W2_2 = W2_1 + 64 * 128;
    _Float16* W3_1 = W2_2 + 64 * 64;
    _Float16* W3_2 = W3_1 + 64 * 64;
    _Float16* W4_1 = W3_2 + 64 * 64;      // 64x192
    _Float16* W4_2 = W4_1 + 64 * 192;
    _Float16* FWIH = W4_2 + 64 * 64;      // 192x64
    _Float16* FWHH = FWIH + 192 * 64;
    _Float16* RWIH = FWHH + 192 * 64;
    _Float16* RWHH = RWIH + 192 * 64;

    auto cvt = [&](const void* src, _Float16* dst, int rows, int sc, int dc) {
        int tot = rows * dc;
        cvt_w_kernel<<<(tot + 255) / 256, 256, 0, stream>>>((const float*)src, dst, rows, sc, dc);
    };
    cvt(d_in[3],  W1_1, 64, INSZ, 32);
    cvt(d_in[5],  W1_2, 64, 64, 64);
    cvt(d_in[7],  W2_1, 64, 128, 128);
    cvt(d_in[9],  W2_2, 64, 64, 64);
    cvt(d_in[11], W3_1, 64, 64, 64);
    cvt(d_in[13], W3_2, 64, 64, 64);
    cvt(d_in[15], W4_1, 64, 192, 192);
    cvt(d_in[17], W4_2, 64, 64, 64);
    cvt(d_in[19], FWIH, 192, 64, 64);
    cvt(d_in[20], FWHH, 192, 64, 64);
    cvt(d_in[23], RWIH, 192, 64, 64);
    cvt(d_in[24], RWHH, 192, 64, 64);

    hipMemsetAsync(AGG, 0, NODE_E * sizeof(float), stream);

    dim3 blk(128);
    mlp1_kernel<<<dim3(BDIM * NNODES, TDIM / 16), blk, 0, stream>>>(
        inputs, W1_1, b1_1, W1_2, b1_2, X1);
    mlp2_kernel<<<dim3(BDIM * NEDGE, TDIM / 16), blk, 0, stream>>>(
        X1, send, recv, W2_1, b2_1, W2_2, b2_2, X2, AGG);
    mlp3_kernel<<<dim3(BDIM * NNODES, TDIM / 16), blk, 0, stream>>>(
        AGG, W3_1, b3_1, W3_2, b3_2, X3);
    mlp4_kernel<<<dim3(BDIM * NEDGE, TDIM / 16), blk, 0, stream>>>(
        X3, X2, send, recv, W4_1, b4_1, W4_2, b4_2, XF);

    float* prior = (float*)d_out;
    float* post  = prior + (size_t)BDIM * TDIM * NEDGE * KOUT;
    float* hT    = post  + (size_t)BDIM * TDIM * NEDGE * KOUT;

    gru_fwd_kernel<<<NSEQ / 16, 32, 0, stream>>>(
        XF, FWIH, FWHH, f_bih, f_bhh, pri_w, pri_b, FY, prior, hT);
    gru_rev_kernel<<<NSEQ / 16, 32, 0, stream>>>(
        XF, RWIH, RWHH, r_bih, r_bhh, enc_w, enc_b, FY, post);
}
